// GATNet_47278999994382
// MI455X (gfx1250) — matrix operand order repaired and checked by reading, not compiled, not run on hardware
//
#include <hip/hip_runtime.h>

// ---------------------------------------------------------------------------
// GATNet on MI455X (gfx1250, wave32, WMMA).
// Dense projections + FC layers run on v_wmma_f32_16x16x32_bf16.
// Register blocking: one wave owns a 16x64 C tile -> A loaded once per k-step
// and reused across 4 WMMAs (4x fewer A fetches, 4 wmma per 10 b128 loads).
// Attention softmax / message scatter / pooling are memory-bound and run on
// VALU + float atomics.
// Workspace requirement: ~640 MB.
// ---------------------------------------------------------------------------

typedef __attribute__((ext_vector_type(16))) __bf16 bf16x16;
typedef __attribute__((ext_vector_type(8)))  __bf16 bf16x8;
typedef __attribute__((ext_vector_type(8)))  float  f32x8;

__device__ __forceinline__ unsigned short f32_to_bf16(float f) {
  unsigned int u = __float_as_uint(f);
  u += 0x7FFFu + ((u >> 16) & 1u);          // round to nearest even
  return (unsigned short)(u >> 16);
}

__device__ __forceinline__ void atomicMaxF(float* addr, float val) {
  unsigned int* ua = (unsigned int*)addr;
  unsigned int old = __float_as_uint(*addr);
  while (__uint_as_float(old) < val) {
    unsigned int prev = atomicCAS(ua, old, __float_as_uint(val));
    if (prev == old) break;
    old = prev;
  }
}

// ----------------------------- fill ----------------------------------------
__global__ void fill_f32(float* __restrict__ p, float v, long long n) {
  long long i = blockIdx.x * (long long)blockDim.x + threadIdx.x;
  long long s = gridDim.x * (long long)blockDim.x;
  for (; i < n; i += s) p[i] = v;
}

// ------------------- f32 -> bf16 packing (zero-padded K) -------------------
__global__ void pack_rows_bf16(const float* __restrict__ src,
                               unsigned short* __restrict__ dst,
                               int M, int K, int Kp) {
  long long total = (long long)M * Kp;
  long long i = blockIdx.x * (long long)blockDim.x + threadIdx.x;
  long long s = gridDim.x * (long long)blockDim.x;
  for (; i < total; i += s) {
    int m = (int)(i / Kp), k = (int)(i % Kp);
    float v = (k < K) ? src[(long long)m * K + k] : 0.0f;
    dst[i] = f32_to_bf16(v);
  }
}

// W [K,N] f32  ->  Wt [Np,Kp] bf16 (transposed, zero-padded)
__global__ void pack_w_transpose_bf16(const float* __restrict__ W,
                                      unsigned short* __restrict__ dst,
                                      int K, int N, int Kp, int Np) {
  long long total = (long long)Np * Kp;
  long long i = blockIdx.x * (long long)blockDim.x + threadIdx.x;
  long long s = gridDim.x * (long long)blockDim.x;
  for (; i < total; i += s) {
    int n = (int)(i / Kp), k = (int)(i % Kp);
    float v = (n < N && k < K) ? W[(long long)k * N + n] : 0.0f;
    dst[i] = f32_to_bf16(v);
  }
}

// ------------------------- WMMA bf16 GEMM ----------------------------------
// C[M,N] = A[M,Kp](bf16) * Bt[Np,Kp]^T(bf16)  (+bias)(+relu)
// One wave per 16x64 tile: A held in VGPRs, 4 accumulators.
// M multiple of 16, Np multiple of 64, Kp multiple of 32.
__global__ void __launch_bounds__(128)
wmma_gemm_bf16(const unsigned short* __restrict__ A,
               const unsigned short* __restrict__ Bt,
               float* __restrict__ C, const float* __restrict__ bias,
               int M, int N, int Kp, int tilesM, int tilesN64, int relu) {
  int wave = blockIdx.x * (blockDim.x >> 5) + (threadIdx.x >> 5);
  if (wave >= tilesM * tilesN64) return;          // wave-uniform branch
  int tm = wave / tilesN64, tn = wave % tilesN64;
  int lane = threadIdx.x & 31;
  int half = lane >> 4;                           // 0: lanes 0-15, 1: 16-31
  int l16  = lane & 15;

  const __bf16* Ab = (const __bf16*)A;
  const __bf16* Bb = (const __bf16*)Bt;
  const __bf16* arow = Ab + (size_t)(tm * 16 + l16) * Kp;
  const __bf16* brow[4];
#pragma unroll
  for (int j = 0; j < 4; ++j)
    brow[j] = Bb + (size_t)(tn * 64 + j * 16 + l16) * Kp;

  f32x8 acc[4] = {};
  for (int k0 = 0; k0 < Kp; k0 += 32) {
    if (k0 + 32 < Kp) {                           // next k-step prefetch
      __builtin_prefetch(arow + k0 + 32, 0, 1);
#pragma unroll
      for (int j = 0; j < 4; ++j) __builtin_prefetch(brow[j] + k0 + 32, 0, 1);
    }
    // A 16x32 bf16: lane half selects K sub-range; two contiguous 8-elem runs
    bf16x8 alo = *(const bf16x8*)(arow + k0 + half * 8);
    bf16x8 ahi = *(const bf16x8*)(arow + k0 + 16 + half * 8);
    bf16x16 a;
#pragma unroll
    for (int i = 0; i < 8; ++i) { a[i] = alo[i]; a[8 + i] = ahi[i]; }
#pragma unroll
    for (int j = 0; j < 4; ++j) {
      // B 32x16 bf16 (transposed weights): lane half selects K sub-range; two contiguous 8-elem runs
      bf16x8 blo = *(const bf16x8*)(brow[j] + k0 + half * 8);
      bf16x8 bhi = *(const bf16x8*)(brow[j] + k0 + 16 + half * 8);
      bf16x16 b;
#pragma unroll
      for (int i = 0; i < 8; ++i) { b[i] = blo[i]; b[8 + i] = bhi[i]; }
      acc[j] = __builtin_amdgcn_wmma_f32_16x16x32_bf16(
          /*neg_a=*/false, a, /*neg_b=*/false, b,
          /*c_mod=*/(short)0, acc[j], /*reuse_a=*/false, /*reuse_b=*/false);
    }
  }

#pragma unroll
  for (int j = 0; j < 4; ++j) {
    int n = tn * 64 + j * 16 + l16;
#pragma unroll
    for (int r = 0; r < 8; ++r) {
      int m = tm * 16 + (half ? 8 + r : r);
      if (m < M && n < N) {
        float v = acc[j][r];
        if (bias) v += bias[n];
        if (relu) v = fmaxf(v, 0.0f);
        C[(size_t)m * N + n] = v;
      }
    }
  }
}

// ----------------------- attention score reduction -------------------------
// a_src[n,h] = sum_c h[n,h,c]*att_src[h,c]; same for dst.
__global__ void attn_scores(const float* __restrict__ hf,
                            const float* __restrict__ att_src,
                            const float* __restrict__ att_dst,
                            float* __restrict__ a_src, float* __restrict__ a_dst,
                            int Nn, int H, int C) {
  int idx = blockIdx.x * blockDim.x + threadIdx.x;
  if (idx >= Nn * H) return;
  int n = idx / H, h = idx % H;
  const float* hp = hf + (size_t)n * H * C + (size_t)h * C;
  const float* ws = att_src + (size_t)h * C;
  const float* wd = att_dst + (size_t)h * C;
  float s = 0.f, d = 0.f;
  for (int c = 0; c < C; ++c) { float v = hp[c]; s += v * ws[c]; d += v * wd[c]; }
  a_src[idx] = s;
  a_dst[idx] = d;
}

// --------------------- edge score + segment max ----------------------------
__global__ void edge_score_max(const int* __restrict__ ei, int E, int Nn, int H,
                               const float* __restrict__ a_src,
                               const float* __restrict__ a_dst,
                               float* __restrict__ ebuf, float* __restrict__ mbuf) {
  long long total = (long long)(E + Nn) * H;
  long long i = blockIdx.x * (long long)blockDim.x + threadIdx.x;
  long long s = gridDim.x * (long long)blockDim.x;
  for (; i < total; i += s) {
    int e = (int)(i / H), h = (int)(i % H);
    int row = (e < E) ? ei[e]     : (e - E);
    int col = (e < E) ? ei[E + e] : (e - E);
    float v = a_src[row * H + h] + a_dst[col * H + h];
    v = (v > 0.f) ? v : 0.2f * v;                 // leaky_relu
    ebuf[i] = v;
    atomicMaxF(&mbuf[col * H + h], v);
  }
}

// ------------------- exp(e - max) + segment sum ----------------------------
__global__ void edge_exp_sum(const int* __restrict__ ei, int E, int Nn, int H,
                             float* __restrict__ ebuf,
                             const float* __restrict__ mbuf,
                             float* __restrict__ denom) {
  long long total = (long long)(E + Nn) * H;
  long long i = blockIdx.x * (long long)blockDim.x + threadIdx.x;
  long long s = gridDim.x * (long long)blockDim.x;
  for (; i < total; i += s) {
    int e = (int)(i / H), h = (int)(i % H);
    int col = (e < E) ? ei[E + e] : (e - E);
    float ex = __expf(ebuf[i] - mbuf[col * H + h]);
    ebuf[i] = ex;
    atomicAdd(&denom[col * H + h], ex);
  }
}

// ------------------- weighted message scatter-add --------------------------
__global__ void edge_message(const int* __restrict__ ei, int E, int Nn, int H, int C,
                             const float* __restrict__ hf,
                             const float* __restrict__ ebuf,
                             const float* __restrict__ denom,
                             float* __restrict__ out) {
  int HC = H * C;
  long long total = (long long)(E + Nn) * HC;
  long long i = blockIdx.x * (long long)blockDim.x + threadIdx.x;
  long long s = gridDim.x * (long long)blockDim.x;
  for (; i < total; i += s) {
    long long e = i / HC;
    int hc = (int)(i % HC);
    int h = hc / C;
    int ee = (int)e;
    int row = (ee < E) ? ei[ee]     : (ee - E);
    int col = (ee < E) ? ei[E + ee] : (ee - E);
    float alpha = ebuf[e * H + h] / (denom[col * H + h] + 1e-16f);
    atomicAdd(&out[(size_t)col * HC + hc], hf[(size_t)row * HC + hc] * alpha);
  }
}

// ------------------------- bias + relu -------------------------------------
__global__ void bias_relu(float* __restrict__ x, const float* __restrict__ bias,
                          int Nn, int F) {
  long long total = (long long)Nn * F;
  long long i = blockIdx.x * (long long)blockDim.x + threadIdx.x;
  long long s = gridDim.x * (long long)blockDim.x;
  for (; i < total; i += s) {
    int f = (int)(i % F);
    x[i] = fmaxf(x[i] + bias[f], 0.0f);
  }
}

// -------------------- global max pool over graphs --------------------------
__global__ void pool_max(const float* __restrict__ hf, const int* __restrict__ batch,
                         float* __restrict__ g, int Nn, int F) {
  long long total = (long long)Nn * F;
  long long i = blockIdx.x * (long long)blockDim.x + threadIdx.x;
  long long s = gridDim.x * (long long)blockDim.x;
  for (; i < total; i += s) {
    int n = (int)(i / F), f = (int)(i % F);
    atomicMaxF(&g[(size_t)batch[n] * F + f], hf[i]);
  }
}

// ===========================================================================
extern "C" void kernel_launch(void* const* d_in, const int* in_sizes, int n_in,
                              void* d_out, int out_size, void* d_ws, size_t ws_size,
                              hipStream_t stream) {
  (void)in_sizes; (void)n_in; (void)out_size; (void)ws_size;

  const float* x     = (const float*)d_in[0];
  const int*   ei    = (const int*)d_in[1];
  const int*   batch = (const int*)d_in[2];
  const float* W1  = (const float*)d_in[3];
  const float* as1 = (const float*)d_in[4];
  const float* ad1 = (const float*)d_in[5];
  const float* b1  = (const float*)d_in[6];
  const float* W2  = (const float*)d_in[7];
  const float* as2 = (const float*)d_in[8];
  const float* ad2 = (const float*)d_in[9];
  const float* b2  = (const float*)d_in[10];
  const float* W3  = (const float*)d_in[11];
  const float* as3 = (const float*)d_in[12];
  const float* ad3 = (const float*)d_in[13];
  const float* b3  = (const float*)d_in[14];
  const float* fc1w = (const float*)d_in[15];
  const float* fc1b = (const float*)d_in[16];
  const float* fc2w = (const float*)d_in[17];
  const float* fc2b = (const float*)d_in[18];
  float* out = (float*)d_out;

  constexpr int Nn = 50000, Ee = 200000, G = 512;
  constexpr int FMAX = 1248;   // max feature width

  // ---------------- workspace layout (256B aligned slabs) ------------------
  size_t off = 0;
  auto slab = [&](size_t bytes) {
    size_t o = off; off += (bytes + 255) & ~(size_t)255; return o;
  };
  char* base = (char*)d_ws;
  float*          hA    = (float*)(base + slab((size_t)Nn * FMAX * 4));   // GEMM out h
  float*          hB    = (float*)(base + slab((size_t)Nn * FMAX * 4));   // layer in/out
  unsigned short* Xbf   = (unsigned short*)(base + slab((size_t)Nn * FMAX * 2));
  unsigned short* Wtbf  = (unsigned short*)(base + slab((size_t)1280 * FMAX * 2));
  float*          asrc  = (float*)(base + slab((size_t)Nn * 4 * 4));
  float*          adst  = (float*)(base + slab((size_t)Nn * 4 * 4));
  float*          mbuf  = (float*)(base + slab((size_t)Nn * 4 * 4));
  float*          denom = (float*)(base + slab((size_t)Nn * 4 * 4));
  float*          ebuf  = (float*)(base + slab((size_t)(Ee + Nn) * 4 * 4));
  float*          g     = (float*)(base + slab((size_t)G * FMAX * 4));
  float*          t1    = (float*)(base + slab((size_t)G * 1024 * 4));

  auto blocks1d = [](long long n, int t) {
    long long b = (n + t - 1) / t;
    if (b > 524288) b = 524288;
    if (b < 1) b = 1;
    return (unsigned)b;
  };
  auto fill = [&](float* p, float v, long long n) {
    fill_f32<<<blocks1d(n, 256), 256, 0, stream>>>(p, v, n);
  };
  // Np must be a multiple of 64 (16x64 wave tiles).
  auto gemm = [&](const unsigned short* A, const unsigned short* Bt, float* C,
                  const float* bias, int M, int N, int Kp, int Np, int relu) {
    int tilesM = M / 16, tilesN64 = Np / 64;
    int waves = tilesM * tilesN64;
    unsigned blocks = (unsigned)((waves + 3) / 4);      // 4 waves / 128-thr block
    wmma_gemm_bf16<<<blocks, 128, 0, stream>>>(A, Bt, C, bias, M, N, Kp,
                                               tilesM, tilesN64, relu);
  };

  // ---------------- one GAT layer ------------------------------------------
  auto gat_layer = [&](const float* xin, int Cin, int H,
                       const float* W, const float* as_, const float* ad_,
                       const float* bias) {
    int Nh = H * Cin;                         // W: [Cin, H*Cin]
    int Kp = ((Cin + 31) / 32) * 32;
    int Np = ((Nh + 63) / 64) * 64;
    pack_rows_bf16<<<blocks1d((long long)Nn * Kp, 256), 256, 0, stream>>>(
        xin, Xbf, Nn, Cin, Kp);
    pack_w_transpose_bf16<<<blocks1d((long long)Np * Kp, 256), 256, 0, stream>>>(
        W, Wtbf, Cin, Nh, Kp, Np);
    gemm(Xbf, Wtbf, hA, nullptr, Nn, Nh, Kp, Np, 0);          // h = x @ W
    attn_scores<<<blocks1d((long long)Nn * H, 256), 256, 0, stream>>>(
        hA, as_, ad_, asrc, adst, Nn, H, Cin);
    fill(mbuf, -3.4e38f, (long long)Nn * H);
    fill(denom, 0.0f, (long long)Nn * H);
    long long eh = (long long)(Ee + Nn) * H;
    edge_score_max<<<blocks1d(eh, 256), 256, 0, stream>>>(
        ei, Ee, Nn, H, asrc, adst, ebuf, mbuf);
    edge_exp_sum<<<blocks1d(eh, 256), 256, 0, stream>>>(
        ei, Ee, Nn, H, ebuf, mbuf, denom);
    fill(hB, 0.0f, (long long)Nn * Nh);                        // xin is consumed
    edge_message<<<blocks1d((long long)(Ee + Nn) * Nh, 256), 256, 0, stream>>>(
        ei, Ee, Nn, H, Cin, hA, ebuf, denom, hB);
    bias_relu<<<blocks1d((long long)Nn * Nh, 256), 256, 0, stream>>>(
        hB, bias, Nn, Nh);
  };

  // conv1: 78 -> 4x78 ; conv2: 312 -> 4x312 ; conv3: 1248 -> 1x1248
  gat_layer(x,  78,   4, W1, as1, ad1, b1);
  gat_layer(hB, 312,  4, W2, as2, ad2, b2);
  gat_layer(hB, 1248, 1, W3, as3, ad3, b3);

  // global max pool (post-ReLU features >= 0, every segment populated)
  fill(g, 0.0f, (long long)G * FMAX);
  pool_max<<<blocks1d((long long)Nn * FMAX, 256), 256, 0, stream>>>(
      hB, batch, g, Nn, FMAX);

  // fc1: [512,1248] @ [1248,1024] + b, relu
  pack_rows_bf16<<<blocks1d((long long)G * 1248, 256), 256, 0, stream>>>(
      g, Xbf, G, 1248, 1248);
  pack_w_transpose_bf16<<<blocks1d((long long)1024 * 1248, 256), 256, 0, stream>>>(
      fc1w, Wtbf, 1248, 1024, 1248, 1024);
  gemm(Xbf, Wtbf, t1, fc1b, G, 1024, 1248, 1024, 1);

  // fc2: [512,1024] @ [1024,128] + b, relu -> d_out
  pack_rows_bf16<<<blocks1d((long long)G * 1024, 256), 256, 0, stream>>>(
      t1, Xbf, G, 1024, 1024);
  pack_w_transpose_bf16<<<blocks1d((long long)128 * 1024, 256), 256, 0, stream>>>(
      fc2w, Wtbf, 1024, 128, 1024, 128);
  gemm(Xbf, Wtbf, out, fc2b, G, 128, 1024, 128, 1);
}
